// MY_SNN_MK1_25941602467943
// MI455X (gfx1250) — matrix operand order repaired and checked by reading, not compiled, not run on hardware
//
#include <hip/hip_runtime.h>
#include <hip/hip_bf16.h>

// ---------- problem constants ----------
#define T_STEPS 8
#define BATCH   64
#define CIN     3
#define CMID    64
#define HW      32
#define HWP     34                      // padded spatial extent (1-halo)
#define NCLS    10
#define PIX     (HW*HW)                 // 1024
#define NPIX    (T_STEPS*BATCH*PIX)     // 524288 output pixels per conv
#define K2      (CMID*9)                // 576 = conv2 im2col K
#define KSTEPS2 (K2/32)                 // 18
#define FCKSTEPS (CMID*PIX/32)          // 2048 FC K-steps of 32
#define V_THR   1.2f

typedef __attribute__((ext_vector_type(16))) _Float16 v16h;
typedef __attribute__((ext_vector_type(8)))  float    v8f;
typedef __attribute__((ext_vector_type(4)))  unsigned int u32x4;

union HalfPack8  { u32x4 u; _Float16 h[8]; };
union HalfPack16 { u32x4 u[2]; v16h v; _Float16 h[16]; };

// s1 padded halo layout: s1[((tb*34 + h+1)*34 + w+1)*64 + c]
#define S1_HALFS ((size_t)T_STEPS*BATCH*HWP*HWP*CMID)        // 37,879,808

// ---------------------------------------------------------------------------
// Kernel 0a: zero the padded spike-1 buffer (halo must be 0 every launch).
// ---------------------------------------------------------------------------
__global__ void snn_zero_s1(_Float16* __restrict__ s1) {
    size_t q = (size_t)blockIdx.x * blockDim.x + threadIdx.x;  // one u32x4 each
    if (q * 8 >= S1_HALFS) return;
    u32x4 z = {0u, 0u, 0u, 0u};
    *(u32x4*)(s1 + q * 8) = z;
}

// ---------------------------------------------------------------------------
// Kernel 0b: pack conv2 weights into WMMA B fragments (contiguous k order, same as A).
//   K = ks*32 + (lane>>4)*16 + e,  N = nt*16 + (lane&15), cin=K&63, tap=K>>6
// ---------------------------------------------------------------------------
__global__ void snn_pack_w2(const float* __restrict__ w2, _Float16* __restrict__ wBfrag) {
    int idx = blockIdx.x * blockDim.x + threadIdx.x;           // 18*4*32*16 = 36864
    if (idx >= KSTEPS2 * 4 * 32 * 16) return;
    int e    = idx & 15;
    int lane = (idx >> 4) & 31;
    int nt   = (idx >> 9) & 3;
    int ks   = idx >> 11;
    int K = ks * 32 + (lane >> 4) * 16 + e;
    int N = nt * 16 + (lane & 15);
    int cin = K & 63, tap = K >> 6;
    wBfrag[idx] = (_Float16)w2[((size_t)N * 64 + cin) * 9 + tap];
}

// ---------------------------------------------------------------------------
// Kernel 0c: pack FC weights into WMMA B fragments (N=10 padded to 16).
//   wfcfrag[(ks*32 + lane)*16 + e] ; K = ks*32 + (lane>>4)*16 + e, N = lane&15
// ---------------------------------------------------------------------------
__global__ void snn_pack_wfc(const float* __restrict__ wfc, _Float16* __restrict__ wfcfrag) {
    int idx = blockIdx.x * blockDim.x + threadIdx.x;           // 2048*32*16 = 1048576
    if (idx >= FCKSTEPS * 32 * 16) return;
    int e    = idx & 15;
    int lane = (idx >> 4) & 31;
    int ks   = idx >> 9;
    int K = ks * 32 + (lane >> 4) * 16 + e;
    int N = lane & 15;
    wfcfrag[idx] = (N < NCLS) ? (_Float16)wfc[((size_t)N << 16) + K] : (_Float16)0.0f;
}

// ---------------------------------------------------------------------------
// Kernel 1: conv1 (3->64, 3x3, pad 1) + bias + LIF threshold.
// Spikes stored f16, padded NHWC (im2col-friendly, halo pre-zeroed).
// ---------------------------------------------------------------------------
__global__ void snn_conv1_lif(const float* __restrict__ x,
                              const float* __restrict__ w1,
                              const float* __restrict__ b1,
                              _Float16* __restrict__ s1) {
    __shared__ float w1s[CMID * 27];
    __shared__ float b1s[CMID];
    int tid = threadIdx.x;
    for (int j = tid; j < CMID * 27; j += blockDim.x) w1s[j] = w1[j];
    if (tid < CMID) b1s[tid] = b1[tid];
    __syncthreads();

    int p = blockIdx.x * blockDim.x + tid;                     // pixel id
    if (p >= NPIX) return;
    int tb = p >> 10;
    int h  = (p >> 5) & 31;
    int w  = p & 31;

    float acc[CMID];
#pragma unroll
    for (int n = 0; n < CMID; ++n) acc[n] = 0.f;

    for (int cin = 0; cin < CIN; ++cin) {
        const float* xc = x + ((size_t)tb * CIN + cin) * PIX;
        for (int dh = -1; dh <= 1; ++dh) {
            int hh = h + dh;
            if ((unsigned)hh >= (unsigned)HW) continue;
            for (int dw = -1; dw <= 1; ++dw) {
                int ww = w + dw;
                if ((unsigned)ww >= (unsigned)HW) continue;
                float xv = xc[hh * HW + ww];
                int ci = cin * 9 + (dh + 1) * 3 + (dw + 1);
#pragma unroll
                for (int n = 0; n < CMID; ++n)
                    acc[n] = fmaf(xv, w1s[n * 27 + ci], acc[n]);
            }
        }
    }

    _Float16* outp = s1 + (((size_t)tb * HWP + (h + 1)) * HWP + (w + 1)) * CMID;
#pragma unroll
    for (int g = 0; g < CMID / 8; ++g) {
        HalfPack8 o;
#pragma unroll
        for (int r = 0; r < 8; ++r) {
            int n = g * 8 + r;
            o.h[r] = (acc[n] + b1s[n] >= V_THR) ? (_Float16)1.0f : (_Float16)0.0f;
        }
        *(u32x4*)(outp + g * 8) = o.u;
    }
}

// ---------------------------------------------------------------------------
// Kernel 2: conv2 (64->64, 3x3) as implicit GEMM via V_WMMA_F32_16X16X32_F16.
// One wave per 16-pixel M-tile; 4 N-tiles per wave (A reuse x4).
// Halo padding -> unconditional A loads; A double-buffered across ksteps and
// all 4 B fragments hoisted ahead of the WMMA quartet for staggered waits.
// ---------------------------------------------------------------------------
__global__ void __launch_bounds__(256)
snn_conv2_lif_wmma(const _Float16* __restrict__ s1,
                   const _Float16* __restrict__ wBfrag,
                   const float* __restrict__ b2,
                   _Float16* __restrict__ s2) {
    int lane  = threadIdx.x & 31;
    int widx  = threadIdx.x >> 5;
    int mtile = blockIdx.x * 8 + widx;            // 32768 tiles, grid exact
    int laneM = lane & 15;
    int hi    = lane >> 4;

    int tb  = mtile >> 6;
    int rem = mtile & 63;
    int h   = rem >> 1;
    int w0  = (rem & 1) * 16;
    int w   = w0 + laneM;                         // A-row pixel for this lane

    const v16h* Bf = (const v16h*)wBfrag;
    // base of this lane's padded input pixel (h+1, w+1)
    const _Float16* s1p = s1 + (((size_t)tb * HWP + (h + 1)) * HWP + (w + 1)) * CMID;

    // A-fragment loader: lane row = laneM; two groups of 8 consecutive K.
    auto load_A = [&](int ks, HalfPack16& A) {
#pragma unroll
        for (int g = 0; g < 2; ++g) {
            int kbase = ks * 32 + hi * 16 + g * 8;    // 8 consecutive K, one tap
            int tap = kbase >> 6, cin = kbase & 63;
            int dh = tap / 3 - 1, dw = tap % 3 - 1;
            A.u[g] = *(const u32x4*)(s1p + ((dh * HWP + dw) * CMID + cin));
        }
    };

    v8f acc[4];
#pragma unroll
    for (int nt = 0; nt < 4; ++nt) acc[nt] = (v8f)(0.f);

    HalfPack16 Acur, Anext;
    load_A(0, Acur);

#pragma unroll
    for (int ks = 0; ks < KSTEPS2; ++ks) {
        // hoist all 4 B fragments (independent of the WMMAs below)
        v16h b0 = Bf[(ks * 4 + 0) * 32 + lane];
        v16h b1 = Bf[(ks * 4 + 1) * 32 + lane];
        v16h b2f = Bf[(ks * 4 + 2) * 32 + lane];
        v16h b3 = Bf[(ks * 4 + 3) * 32 + lane];
        if (ks + 1 < KSTEPS2) {
            load_A(ks + 1, Anext);                    // double-buffer A
            __builtin_prefetch((const void*)(Bf + ((ks + 1) * 4) * 32 + lane), 0, 3);
        }
        acc[0] = __builtin_amdgcn_wmma_f32_16x16x32_f16(false, Acur.v, false, b0,
                                                        (short)0, acc[0], false, false);
        acc[1] = __builtin_amdgcn_wmma_f32_16x16x32_f16(false, Acur.v, false, b1,
                                                        (short)0, acc[1], false, false);
        acc[2] = __builtin_amdgcn_wmma_f32_16x16x32_f16(false, Acur.v, false, b2f,
                                                        (short)0, acc[2], false, false);
        acc[3] = __builtin_amdgcn_wmma_f32_16x16x32_f16(false, Acur.v, false, b3,
                                                        (short)0, acc[3], false, false);
        Acur = Anext;
    }

    // ---- epilogue: bias + LIF threshold, store [t,b,c,h,w] ----
#pragma unroll
    for (int nt = 0; nt < 4; ++nt) {
        int n = nt * 16 + (lane & 15);
        float bias = b2[n];
        HalfPack8 o;
#pragma unroll
        for (int r = 0; r < 8; ++r)                    // D: M = r + hi*8 -> w
            o.h[r] = (acc[nt][r] + bias >= V_THR) ? (_Float16)1.0f : (_Float16)0.0f;
        size_t base = ((((size_t)tb * CMID + n) * HW + h) * HW + (w0 + hi * 8));
        *(u32x4*)(s2 + base) = o.u;
    }
}

// ---------------------------------------------------------------------------
// Kernel 3: S[b,i] = sum_t spikes2[t,b,i]   (values 0..8, exact in f16)
// ---------------------------------------------------------------------------
__global__ void snn_presum_t(const _Float16* __restrict__ s2, _Float16* __restrict__ S) {
    int q = blockIdx.x * blockDim.x + threadIdx.x;     // 524288 threads, 8 elems each
    if (q >= (BATCH * CMID * PIX) / 8) return;
    int b  = q >> 13;
    int io = (q & 8191) * 8;
    HalfPack8 accp; accp.u = (u32x4){0u, 0u, 0u, 0u};
#pragma unroll
    for (int t = 0; t < T_STEPS; ++t) {
        HalfPack8 v;
        v.u = *(const u32x4*)(s2 + (((size_t)(t * BATCH + b)) << 16) + io);
#pragma unroll
        for (int r = 0; r < 8; ++r) accp.h[r] = accp.h[r] + v.h[r];
    }
    *(u32x4*)(S + ((size_t)b << 16) + io) = accp.u;
}

// ---------------------------------------------------------------------------
// Kernel 4: out[b,o] = T*bfc[o]   (harness poisons d_out every run)
// ---------------------------------------------------------------------------
__global__ void snn_init_out(const float* __restrict__ bfc, float* __restrict__ out) {
    int i = blockIdx.x * blockDim.x + threadIdx.x;
    if (i < BATCH * NCLS) out[i] = (float)T_STEPS * bfc[i % NCLS];
}

// ---------------------------------------------------------------------------
// Kernel 5: FC via WMMA, split-K.  M=64 (batch, 4 tiles), N=10 (pad 16),
// K=65536 split into 64 chunks; pre-packed B fragments; f32 atomics to d_out.
// ---------------------------------------------------------------------------
__global__ void __launch_bounds__(32)
snn_fc_wmma(const _Float16* __restrict__ S,
            const _Float16* __restrict__ wfcfrag,
            float* __restrict__ out) {
    int lane  = threadIdx.x & 31;
    int laneM = lane & 15;
    int hi    = lane >> 4;
    int mt = blockIdx.x & 3;
    int kc = blockIdx.x >> 2;                          // 64 K-chunks of 1024
    int brow = mt * 16 + laneM;
    int ncol = lane & 15;

    const v16h* Bf = (const v16h*)wfcfrag;
    const _Float16* Arow = S + ((size_t)brow << 16);

    v8f acc = (v8f)(0.f);
#pragma unroll
    for (int ks = 0; ks < 32; ++ks) {
        int ksg = kc * 32 + ks;                        // global K-step
        int kb  = ksg * 32;
        HalfPack16 A;
        A.u[0] = *(const u32x4*)(Arow + kb + hi * 16);
        A.u[1] = *(const u32x4*)(Arow + kb + hi * 16 + 8);
        v16h b = Bf[ksg * 32 + lane];                  // coalesced 32B/lane
        acc = __builtin_amdgcn_wmma_f32_16x16x32_f16(
            false, A.v, false, b, (short)0, acc, false, false);
    }
    if (ncol < NCLS) {
#pragma unroll
        for (int r = 0; r < 8; ++r) {
            int b = mt * 16 + r + hi * 8;
            atomicAdd(&out[b * NCLS + ncol], acc[r]);
        }
    }
}

// ---------------------------------------------------------------------------
extern "C" void kernel_launch(void* const* d_in, const int* in_sizes, int n_in,
                              void* d_out, int out_size, void* d_ws, size_t ws_size,
                              hipStream_t stream) {
    const float* x   = (const float*)d_in[0];   // [8,64,3,32,32]
    const float* w1  = (const float*)d_in[1];   // [64,3,3,3]
    const float* b1  = (const float*)d_in[2];   // [64]
    const float* w2  = (const float*)d_in[3];   // [64,64,3,3]
    const float* b2  = (const float*)d_in[4];   // [64]
    const float* wfc = (const float*)d_in[5];   // [10,65536]
    const float* bfc = (const float*)d_in[6];   // [10]
    float* out = (float*)d_out;                 // [64,10]

    char* ws = (char*)d_ws;
    const size_t S1_BYTES  = S1_HALFS * 2;                     // 75.8 MB padded spikes1
    const size_t S2_BYTES  = (size_t)NPIX * CMID * 2;          // 67 MB spikes2 (tbchw)
    const size_t SS_BYTES  = (size_t)BATCH * CMID * PIX * 2;   // 8.4 MB t-summed spikes
    const size_t WB_BYTES  = (size_t)KSTEPS2 * 4 * 32 * 16 * 2;// 73.7 KB conv2 B frags
    _Float16* s1      = (_Float16*)(ws);
    _Float16* s2      = (_Float16*)(ws + S1_BYTES);
    _Float16* Ssum    = (_Float16*)(ws + S1_BYTES + S2_BYTES);
    _Float16* wBfrag  = (_Float16*)(ws + S1_BYTES + S2_BYTES + SS_BYTES);
    _Float16* wfcfrag = (_Float16*)(ws + S1_BYTES + S2_BYTES + SS_BYTES + WB_BYTES);

    snn_zero_s1   <<<(int)((S1_HALFS/8 + 255)/256), 256, 0, stream>>>(s1);
    snn_pack_w2   <<<(KSTEPS2*4*32*16 + 255)/256, 256, 0, stream>>>(w2, wBfrag);
    snn_pack_wfc  <<<(FCKSTEPS*32*16 + 255)/256, 256, 0, stream>>>(wfc, wfcfrag);
    snn_conv1_lif <<<NPIX/256, 256, 0, stream>>>(x, w1, b1, s1);
    snn_conv2_lif_wmma<<<(NPIX/16)/8, 256, 0, stream>>>(s1, wBfrag, b2, s2);
    snn_presum_t  <<<(BATCH*CMID*PIX/8)/256, 256, 0, stream>>>(s2, Ssum);
    snn_init_out  <<<3, 256, 0, stream>>>(bfc, out);
    snn_fc_wmma   <<<4*64, 32, 0, stream>>>(Ssum, wfcfrag, out);
}
